// CT_UNet_54357106098602
// MI455X (gfx1250) — matrix operand order repaired and checked by reading, not compiled, not run on hardware
//
#include <hip/hip_runtime.h>
#include <hip/hip_bf16.h>
#include <math.h>

// ---------------------------------------------------------------------------
// CT-UNet + Neural CDE for MI455X (gfx1250, wave32, WMMA).
// 3x3 convs = implicit GEMM on v_wmma_f32_16x16x32_f16 (f32 accumulate).
// K is permuted TAP-MAJOR (K' = rs*cinT + ci); kernel iterates rs outer /
// channel-chunks inner, so the hot loop is: 16 immediate-offset loads (W is
// a template constant), 16 cvt+select, COARS packed-A b128 loads, COARS
// WMMAs, two pointer bumps.  Weights prepacked on device into per-lane
// A-fragment layout in K' order.  Fused concat + bias + activation.
// ---------------------------------------------------------------------------

typedef __attribute__((ext_vector_type(16))) _Float16 v16h;
typedef __attribute__((ext_vector_type(8)))  float    v8f;

__device__ __forceinline__ float act_apply(float x, int act) {
    if (act == 1) return fmaxf(x, 0.0f);             // relu
    if (act == 2) return tanhf(x);                   // tanh
    if (act == 3) return 1.0f / (1.0f + __expf(-x)); // sigmoid
    return x;
}

// ---------------------------------------------------------------------------
// Weight prepack: w (cout, cinT, 3, 3) f32 -> packed f16 A-fragments, K' order
// layout: [coTile][kStep][lane][16]; element e of lane's v16h holds
//   K' = kStep*32 + 16*(lane>>4) + e,  M(row) = (lane&15)
// ---------------------------------------------------------------------------
__global__ void pack_w_k(const float* __restrict__ w, _Float16* __restrict__ pw,
                         long n, int cinT, int Ktot, int nK)
{
    long idx = (long)blockIdx.x * blockDim.x + threadIdx.x;
    if (idx >= n) return;
    int e    = (int)(idx & 15);
    int lane = (int)((idx >> 4) & 31);
    long t   = idx >> 9;
    int ks   = (int)(t % nK);
    int ct   = (int)(t / nK);
    int half = lane >> 4, l16 = lane & 15;
    int co   = ct * 16 + l16;
    int ka   = ks * 32 + (half << 4) + e;   // K' index
    float v  = 0.0f;
    if (ka < Ktot) {
        int rs = ka / cinT;          // tap
        int ci = ka - rs * cinT;     // channel
        v = w[((long)co * cinT + ci) * 9 + rs];
    }
    pw[idx] = (_Float16)v;
}

// ---------------------------------------------------------------------------
// Implicit-GEMM 3x3 SAME conv, NCHW, H == W == template constant.
// One wave computes (16*COARS) output channels x 16 pixels.
// ---------------------------------------------------------------------------
template <int COARS, int W>
__global__ void conv3x3_wmma(const float* __restrict__ inA, int cinA,
                             const float* __restrict__ inB, int cinB,
                             const _Float16* __restrict__ pw,
                             const float* __restrict__ bias,
                             float* __restrict__ out,
                             int cout, int act, int nK)
{
    constexpr int H  = W;
    constexpr int HW = W * W;

    const int lane = threadIdx.x;      // 0..31
    const int half = lane >> 4;
    const int l16  = lane & 15;
    const int pxBase = blockIdx.x << 4;
    const int ctBase = blockIdx.y * COARS;
    const int img    = blockIdx.z;
    const int cinT   = cinA + cinB;

    const float* __restrict__ srcA = inA + (size_t)img * cinA * HW;
    const float* __restrict__ srcB = inB ? (inB + (size_t)img * cinB * HW) : srcA;

    // B-fragment pixel owned by this lane (column N of the GEMM)
    const int p  = pxBase + l16;
    const int py = p / W;              // constant divide -> shift
    const int px = p - py * W;

    // per-lane tap validity bits, branchless: 3 row bits x 3 col bits
    const unsigned rok[3] = { (unsigned)(py - 1) < (unsigned)H, 1u,
                              (unsigned)(py + 1) < (unsigned)H };
    const unsigned cok[3] = { (unsigned)(px - 1) < (unsigned)W, 1u,
                              (unsigned)(px + 1) < (unsigned)W };
    unsigned vmask = 0;
#pragma unroll
    for (int kh = 0; kh < 3; ++kh)
#pragma unroll
        for (int kw = 0; kw < 3; ++kw)
            vmask |= (rok[kh] & cok[kw]) << (kh * 3 + kw);

    // packed-A cursor: advances 32 v16h per K-step; tile c adds c*nK*32
    const v16h* ap = (const v16h*)pw + (size_t)ctBase * nK * 32 + lane;

    v8f acc[COARS];
#pragma unroll
    for (int c = 0; c < COARS; ++c) acc[c] = (v8f){};

    if (cinT >= 32) {
        // =========== main path: rs outer, 32-channel chunks inner ===========
        const int nA32 = cinA >> 5;
        const int nB32 = cinB >> 5;
        for (int rs = 0; rs < 9; ++rs) {
            const int kh  = (rs * 11) >> 5;              // rs/3
            const int kw  = rs - 3 * kh;
            const int off = (kh - 1) * W + (kw - 1);
            const unsigned vOK = (vmask >> rs) & 1u;
            const int o = vOK ? off : 0;                 // clamped, in-range

            // inner K-step body (16 immediate-offset loads + COARS WMMAs)
            auto kstep = [&](const float* bp) {
                float ldv[16];
#pragma unroll
                for (int i = 0; i < 16; ++i)
                    ldv[i] = bp[i * HW];                 // constant offsets
                v16h bfrag;
#pragma unroll
                for (int i = 0; i < 16; ++i)
                    bfrag[i] = (_Float16)(vOK ? ldv[i] : 0.0f);
#pragma unroll
                for (int c = 0; c < COARS; ++c) {
                    v16h afrag = ap[(size_t)c * nK * 32];
                    acc[c] = __builtin_amdgcn_wmma_f32_16x16x32_f16(
                                 false, afrag, false, bfrag, (short)0, acc[c],
                                 false, false);
                }
                __builtin_prefetch(ap + 32, 0, 1);       // next packed stripe
                ap += 32;
            };

            const float* bpA = srcA + (size_t)(half << 4) * HW + (p + o);
            for (int s = 0; s < nA32; ++s) { kstep(bpA); bpA += (size_t)32 * HW; }
            const float* bpB = srcB + (size_t)(half << 4) * HW + (p + o);
            for (int s = 0; s < nB32; ++s) { kstep(bpB); bpB += (size_t)32 * HW; }
        }
    } else {
        // =========== fallback: cinT == 1 (Ktot = 9, nK = 1) =================
        v16h bfrag;
#pragma unroll
        for (int i = 0; i < 16; ++i) {
            int rs = (half << 4) + i;                    // K' = rs directly
            int rsc = (rs < 8) ? rs : 8;
            int kh  = (rsc * 11) >> 5;
            int off = (kh - 1) * W + (rsc - 3 * kh - 1);
            unsigned v = (rs < 9) ? ((vmask >> rs) & 1u) : 0u;
            int o = v ? off : 0;
            float ld = srcA[p + o];
            bfrag[i] = (_Float16)(v ? ld : 0.0f);
        }
#pragma unroll
        for (int c = 0; c < COARS; ++c) {
            v16h afrag = ap[(size_t)c * nK * 32];
            acc[c] = __builtin_amdgcn_wmma_f32_16x16x32_f16(
                         false, afrag, false, bfrag, (short)0, acc[c], false, false);
        }
    }

    // Epilogue: C/D layout -> lane&15 = N (pixel), vgpr r -> M = r + 8*half
#pragma unroll
    for (int c = 0; c < COARS; ++c) {
#pragma unroll
        for (int r = 0; r < 8; ++r) {
            int oc = (ctBase + c) * 16 + r + (half << 3);
            float v = acc[c][r] + bias[oc];
            out[((size_t)img * cout + oc) * HW + p] = act_apply(v, act);
        }
    }
}

// ---------------------------------------------------------------------------
// 2x2 max pool, stride 2
// ---------------------------------------------------------------------------
__global__ void maxpool2_k(const float* __restrict__ in, float* __restrict__ out,
                           long n, int H, int W)
{
    long idx = (long)blockIdx.x * blockDim.x + threadIdx.x;
    if (idx >= n) return;
    int oW = W >> 1, oH = H >> 1;
    int ow = (int)(idx % oW);
    long t = idx / oW;
    int oh = (int)(t % oH);
    long c = t / oH;
    const float* p = in + ((long)c * H + 2 * oh) * W + 2 * ow;
    out[idx] = fmaxf(fmaxf(p[0], p[1]), fmaxf(p[W], p[W + 1]));
}

// ---------------------------------------------------------------------------
// Nearest 2x upsample
// ---------------------------------------------------------------------------
__global__ void upsample2_k(const float* __restrict__ in, float* __restrict__ out,
                            long n, int H, int W)
{
    long idx = (long)blockIdx.x * blockDim.x + threadIdx.x;
    if (idx >= n) return;
    int oW = W << 1, oH = H << 1;
    int ow = (int)(idx % oW);
    long t = idx / oW;
    int oh = (int)(t % oH);
    long c = t / oH;
    out[idx] = in[((long)c * H + (oh >> 1)) * W + (ow >> 1)];
}

// ---------------------------------------------------------------------------
// 2x2 stride-2 conv transpose: out[n,d,2h+a,2w+b] = sum_c x[n,c,h,w]*W[c,d,a,b]
// ---------------------------------------------------------------------------
__global__ void convT2_k(const float* __restrict__ in, const float* __restrict__ wgt,
                         const float* __restrict__ bias, float* __restrict__ out,
                         long n, int cin, int cd, int H, int W)
{
    long idx = (long)blockIdx.x * blockDim.x + threadIdx.x;
    if (idx >= n) return;
    int oW = W << 1, oH = H << 1;
    int ow = (int)(idx % oW);
    long t = idx / oW;
    int oh = (int)(t % oH);
    t /= oH;
    int d  = (int)(t % cd);
    long nn = t / cd;
    int h = oh >> 1, a = oh & 1, w2 = ow >> 1, b2 = ow & 1;
    float s = bias[d];
    const float* ip = in + ((long)nn * cin) * H * W + (long)h * W + w2;
    const float* wp = wgt + (long)d * 4 + a * 2 + b2;
    for (int c = 0; c < cin; ++c)
        s += ip[(long)c * H * W] * wp[(long)c * cd * 4];
    out[idx] = s;
}

// ---------------------------------------------------------------------------
// Final 1x1 conv (cin -> 1) + sigmoid
// ---------------------------------------------------------------------------
__global__ void conv1x1_sig_k(const float* __restrict__ in, const float* __restrict__ wgt,
                              const float* __restrict__ bias, float* __restrict__ out,
                              long n, int cin, int HW)
{
    long idx = (long)blockIdx.x * blockDim.x + threadIdx.x;
    if (idx >= n) return;
    int p = (int)(idx % HW);
    long nn = idx / HW;
    float s = bias[0];
    const float* ip = in + (long)nn * cin * HW + p;
    for (int c = 0; c < cin; ++c)
        s += wgt[c] * ip[(long)c * HW];
    out[idx] = 1.0f / (1.0f + __expf(-s));
}

// ---------------------------------------------------------------------------
// Natural cubic spline second derivatives M (Thomas solve, F=8, M0=M7=0)
// ---------------------------------------------------------------------------
__global__ void spline_M_k(const float* __restrict__ lat, const float* __restrict__ times,
                           float* __restrict__ Mout, long n, int F, int CHW)
{
    long idx = (long)blockIdx.x * blockDim.x + threadIdx.x;
    if (idx >= n) return;
    int j = (int)(idx % CHW);
    int b = (int)(idx / CHW);

    float t[8], y[8];
    for (int i = 0; i < 8; ++i) {
        t[i] = times[b * F + i];
        y[i] = lat[((long)(b * F + i)) * CHW + j];
    }
    float h[7], d[7];
    for (int i = 0; i < 7; ++i) { h[i] = t[i + 1] - t[i]; d[i] = (y[i + 1] - y[i]) / h[i]; }

    float cp[8], dp[8], M[8];
    float diag = 2.0f * (h[0] + h[1]);
    cp[1] = h[1] / diag;
    dp[1] = 6.0f * (d[1] - d[0]) / diag;
    for (int i = 2; i <= 6; ++i) {
        float m = 2.0f * (h[i - 1] + h[i]) - h[i - 1] * cp[i - 1];
        cp[i] = h[i] / m;
        dp[i] = (6.0f * (d[i] - d[i - 1]) - h[i - 1] * dp[i - 1]) / m;
    }
    M[0] = 0.0f; M[7] = 0.0f;
    M[6] = dp[6];
    for (int i = 5; i >= 1; --i) M[i] = dp[i] - cp[i] * M[i + 1];
    for (int i = 0; i < 8; ++i) Mout[((long)(b * F + i)) * CHW + j] = M[i];
}

// ---------------------------------------------------------------------------
// CDE init: zcur = z0 ; z_all[:,0] = z0
// ---------------------------------------------------------------------------
__global__ void cde_init_k(const float* __restrict__ z0, float* __restrict__ zcur,
                           float* __restrict__ zall, long n, int F, int CHW)
{
    long idx = (long)blockIdx.x * blockDim.x + threadIdx.x;
    if (idx >= n) return;
    int j = (int)(idx % CHW);
    long b = idx / CHW;
    float v = z0[idx];
    zcur[idx] = v;
    zall[(b * F) * (long)CHW + j] = v;
}

// ---------------------------------------------------------------------------
// RK4 stage: kout = vf * spline_deriv(tau); ztmp = zcur + coef*dt_sub*kout
// ---------------------------------------------------------------------------
__global__ void cde_stage_k(const float* __restrict__ vf, const float* __restrict__ lat,
                            const float* __restrict__ Mb, const float* __restrict__ times,
                            const float* __restrict__ zcur, float* __restrict__ kout,
                            float* __restrict__ ztmp,
                            long n, int itv, int sub, float frac, float coef, int F, int CHW)
{
    long idx = (long)blockIdx.x * blockDim.x + threadIdx.x;
    if (idx >= n) return;
    int j = (int)(idx % CHW);
    int b = (int)(idx / CHW);

    float t0 = times[b * F + itv];
    float t1 = times[b * F + itv + 1];
    float hh = t1 - t0;
    float dts = hh * 0.25f;
    float tau = t0 + ((float)sub + frac) * dts;

    long o0 = ((long)(b * F + itv)) * CHW + j;
    long o1 = o0 + CHW;
    float y0 = lat[o0], y1 = lat[o1];
    float M0 = Mb[o0],  M1 = Mb[o1];

    float a = t1 - tau, c = tau - t0;
    float sd = (-M0 * a * a + M1 * c * c) / (2.0f * hh)
             + (y1 - y0) / hh - (M1 - M0) * hh / 6.0f;

    float kv = vf[idx] * sd;
    kout[idx] = kv;
    if (ztmp) ztmp[idx] = zcur[idx] + coef * dts * kv;
}

// ---------------------------------------------------------------------------
// RK4 combine: z += dt_sub/6*(k1+2k2+2k3+k4); optionally store into z_all
// ---------------------------------------------------------------------------
__global__ void cde_rk4_k(float* __restrict__ zcur,
                          const float* __restrict__ k1, const float* __restrict__ k2,
                          const float* __restrict__ k3, const float* __restrict__ k4,
                          const float* __restrict__ times, float* __restrict__ zall,
                          long n, int itv, int storeIdx, int F, int CHW)
{
    long idx = (long)blockIdx.x * blockDim.x + threadIdx.x;
    if (idx >= n) return;
    int j = (int)(idx % CHW);
    int b = (int)(idx / CHW);
    float t0 = times[b * F + itv];
    float t1 = times[b * F + itv + 1];
    float dts = (t1 - t0) * 0.25f;
    float z = zcur[idx] + dts * (1.0f / 6.0f) *
              (k1[idx] + 2.0f * k2[idx] + 2.0f * k3[idx] + k4[idx]);
    zcur[idx] = z;
    if (storeIdx > 0) zall[((long)(b * F + storeIdx)) * CHW + j] = z;
}

// ===========================================================================
// Host orchestration
// ===========================================================================
static inline dim3 ewgrid(long n) { return dim3((unsigned)((n + 255) / 256)); }

struct PW { const _Float16* p; int nK; };

extern "C" void kernel_launch(void* const* d_in, const int* in_sizes, int n_in,
                              void* d_out, int out_size, void* d_ws, size_t ws_size,
                              hipStream_t stream)
{
    (void)in_sizes; (void)out_size; (void)ws_size;

    const int B = 4, F = 8, NF = 32;
    const int CHW = 256 * 8 * 8;

    const float* x     = (const float*)d_in[0];
    const float* times = (const float*)d_in[1];
    const float* y0    = (const float*)d_in[2];

    // ---- parameter leaves in _make_params insertion order -------------------
    long psz[60]; int pi = 0;
    auto cbsz = [&](long cin, long cout) {
        psz[pi++] = cout * cin * 9; psz[pi++] = cout;
        psz[pi++] = cout * cout * 9; psz[pi++] = cout;
    };
    cbsz(1, 32);  cbsz(32, 64);  cbsz(64, 128);  cbsz(128, 256);   // down1..4  -> 0..15
    cbsz(1, 32);  cbsz(32, 64);  cbsz(64, 128);  cbsz(128, 256);   // me1..4    -> 16..31
    psz[pi++] = 512L * 256 * 9; psz[pi++] = 512;                   // cde_w1/b1 -> 32,33
    psz[pi++] = 256L * 512 * 9; psz[pi++] = 256;                   // cde_w2/b2 -> 34,35
    cbsz(512, 256); cbsz(256, 128); cbsz(128, 64); cbsz(64, 32);   // up1..4    -> 36..51
    psz[pi++] = 256L * 128 * 4; psz[pi++] = 128;                   // ct1       -> 52,53
    psz[pi++] = 128L * 64 * 4;  psz[pi++] = 64;                    // ct2       -> 54,55
    psz[pi++] = 64L * 32 * 4;   psz[pi++] = 32;                    // ct3       -> 56,57
    psz[pi++] = 32;             psz[pi++] = 1;                     // ro        -> 58,59

    const float* P[60];
    if (n_in >= 63) {
        for (int i = 0; i < 60; ++i) P[i] = (const float*)d_in[3 + i];
    } else {
        const float* base = (const float*)d_in[3];
        long off = 0;
        for (int i = 0; i < 60; ++i) { P[i] = base + off; off += psz[i]; }
    }

    // ---- workspace arena (floats) ------------------------------------------
    float* wsp = (float*)d_ws;
    long off = 0;
    auto alloc = [&](long nf) { float* p = wsp + off; off += nf; return p; };
    float* res1 = alloc(16777216);   // (32, 32,128,128)
    float* t1   = alloc(16777216);
    float* p1   = alloc(4194304);    // (32, 32, 64, 64) / ct1o (32,128,32,32)
    float* res2 = alloc(8388608);    // (32, 64, 64, 64)
    float* t2   = alloc(8388608);
    float* p2   = alloc(2097152);    // (32, 64, 32, 32) / xz (32,256,16,16)
    float* res3 = alloc(4194304);    // (32,128, 32, 32)
    float* t3   = alloc(4194304);
    float* p3   = alloc(1048576);    // (32,128, 16, 16)
    float* res4 = alloc(2097152);    // (32,256, 16, 16)
    float* t4   = alloc(2097152);
    float* lat  = alloc(524288);     // (32,256, 8, 8)
    float* Mb   = alloc(524288);
    float* big  = alloc(16777216);   // mask-encoder temps / u4a
    float* z0b  = alloc(65536);
    float* zcur = alloc(65536);
    float* ztmp = alloc(65536);
    float* kb1  = alloc(65536); float* kb2 = alloc(65536);
    float* kb3  = alloc(65536); float* kb4 = alloc(65536);
    float* h1   = alloc(131072);
    float* vf   = alloc(65536);
    float* zall = alloc(524288);
    float* pwRegion = alloc(4194304);          // 8M halves = 16MB packed weights

    // ---- prepack all 3x3 conv weights (tap-major K') ------------------------
    long poff = 0;
    auto pack = [&](const float* wsrc, int cout, int cinT) -> PW {
        int Ktot = cinT * 9;
        int nK   = (Ktot + 31) / 32;
        _Float16* dst = (_Float16*)pwRegion + poff;
        long nh = (long)(cout / 16) * nK * 512;
        poff += nh;
        pack_w_k<<<ewgrid(nh), 256, 0, stream>>>(wsrc, dst, nh, cinT, Ktot, nK);
        return PW{dst, nK};
    };
    PW wd1a = pack(P[0], 32, 1),     wd1b = pack(P[2], 32, 32);
    PW wd2a = pack(P[4], 64, 32),    wd2b = pack(P[6], 64, 64);
    PW wd3a = pack(P[8], 128, 64),   wd3b = pack(P[10], 128, 128);
    PW wd4a = pack(P[12], 256, 128), wd4b = pack(P[14], 256, 256);
    PW wm1a = pack(P[16], 32, 1),    wm1b = pack(P[18], 32, 32);
    PW wm2a = pack(P[20], 64, 32),   wm2b = pack(P[22], 64, 64);
    PW wm3a = pack(P[24], 128, 64),  wm3b = pack(P[26], 128, 128);
    PW wm4a = pack(P[28], 256, 128), wm4b = pack(P[30], 256, 256);
    PW wcde1 = pack(P[32], 512, 256), wcde2 = pack(P[34], 256, 512);
    PW wu1a = pack(P[36], 256, 512), wu1b = pack(P[38], 256, 256);
    PW wu2a = pack(P[40], 128, 256), wu2b = pack(P[42], 128, 128);
    PW wu3a = pack(P[44], 64, 128),  wu3b = pack(P[46], 64, 64);
    PW wu4a = pack(P[48], 32, 64),   wu4b = pack(P[50], 32, 32);

    // ---- conv dispatch: (COARS, W) fixed combos ----------------------------
    auto conv = [&](const float* inA, int cA, const float* inB, int cB,
                    PW w, const float* bi, float* outp,
                    int N, int Wd, int cout, int act) {
        unsigned tilesX = (unsigned)((Wd * Wd) / 16);
        switch (Wd) {
        case 128: {
            dim3 g(tilesX, (unsigned)(cout / 32), (unsigned)N);
            conv3x3_wmma<2, 128><<<g, dim3(32), 0, stream>>>(inA, cA, inB, cB, w.p, bi,
                                                             outp, cout, act, w.nK);
        } break;
        case 64: {
            dim3 g(tilesX, (unsigned)(cout / 64), (unsigned)N);
            conv3x3_wmma<4, 64><<<g, dim3(32), 0, stream>>>(inA, cA, inB, cB, w.p, bi,
                                                            outp, cout, act, w.nK);
        } break;
        case 32: {
            dim3 g(tilesX, (unsigned)(cout / 64), (unsigned)N);
            conv3x3_wmma<4, 32><<<g, dim3(32), 0, stream>>>(inA, cA, inB, cB, w.p, bi,
                                                            outp, cout, act, w.nK);
        } break;
        case 16: {
            dim3 g(tilesX, (unsigned)(cout / 64), (unsigned)N);
            conv3x3_wmma<4, 16><<<g, dim3(32), 0, stream>>>(inA, cA, inB, cB, w.p, bi,
                                                            outp, cout, act, w.nK);
        } break;
        default: { // 8x8 CDE convs: COARS=1 for occupancy
            dim3 g(tilesX, (unsigned)(cout / 16), (unsigned)N);
            conv3x3_wmma<1, 8><<<g, dim3(32), 0, stream>>>(inA, cA, inB, cB, w.p, bi,
                                                           outp, cout, act, w.nK);
        } break;
        }
    };
    auto pool = [&](const float* in, float* outp, long NC, int H, int W) {
        long n = NC * (H / 2) * (W / 2);
        maxpool2_k<<<ewgrid(n), 256, 0, stream>>>(in, outp, n, H, W);
    };

    // ======================= mask encoder (z0) ==============================
    float* meA = big;
    float* meB = big + 2097152;
    conv(y0, 1, 0, 0, wm1a, P[17], meA, B, 128, 32, 1);
    conv(meA, 32, 0, 0, wm1b, P[19], meB, B, 128, 32, 1);
    pool(meB, meA, (long)B * 32, 128, 128);
    conv(meA, 32, 0, 0, wm2a, P[21], meB, B, 64, 64, 1);
    conv(meB, 64, 0, 0, wm2b, P[23], meA, B, 64, 64, 1);
    pool(meA, meB, (long)B * 64, 64, 64);
    conv(meB, 64, 0, 0, wm3a, P[25], meA, B, 32, 128, 1);
    conv(meA, 128, 0, 0, wm3b, P[27], meB, B, 32, 128, 1);
    pool(meB, meA, (long)B * 128, 32, 32);
    conv(meA, 128, 0, 0, wm4a, P[29], meB, B, 16, 256, 1);
    conv(meB, 256, 0, 0, wm4b, P[31], meA, B, 16, 256, 1);
    pool(meA, z0b, (long)B * 256, 16, 16);

    // ======================= down path ======================================
    conv(x, 1, 0, 0, wd1a, P[1], t1, NF, 128, 32, 1);
    conv(t1, 32, 0, 0, wd1b, P[3], res1, NF, 128, 32, 1);
    pool(res1, p1, (long)NF * 32, 128, 128);
    conv(p1, 32, 0, 0, wd2a, P[5], t2, NF, 64, 64, 1);
    conv(t2, 64, 0, 0, wd2b, P[7], res2, NF, 64, 64, 1);
    pool(res2, p2, (long)NF * 64, 64, 64);
    conv(p2, 64, 0, 0, wd3a, P[9], t3, NF, 32, 128, 1);
    conv(t3, 128, 0, 0, wd3b, P[11], res3, NF, 32, 128, 1);
    pool(res3, p3, (long)NF * 128, 32, 32);
    conv(p3, 128, 0, 0, wd4a, P[13], t4, NF, 16, 256, 1);
    conv(t4, 256, 0, 0, wd4b, P[15], res4, NF, 16, 256, 1);
    pool(res4, lat, (long)NF * 256, 16, 16);

    // ======================= CDE solve ======================================
    {
        long n = (long)B * CHW;
        spline_M_k<<<ewgrid(n), 256, 0, stream>>>(lat, times, Mb, n, F, CHW);
        cde_init_k<<<ewgrid(n), 256, 0, stream>>>(z0b, zcur, zall, n, F, CHW);

        const float fracs[4] = {0.0f, 0.5f, 0.5f, 1.0f};
        const float coefs[4] = {0.5f, 0.5f, 1.0f, 0.0f};
        float* kb[4] = {kb1, kb2, kb3, kb4};

        for (int itv = 0; itv < F - 1; ++itv) {
            for (int sub = 0; sub < 4; ++sub) {
                for (int st = 0; st < 4; ++st) {
                    const float* zeval = (st == 0) ? zcur : ztmp;
                    conv(zeval, 256, 0, 0, wcde1, P[33], h1, B, 8, 512, 1);
                    conv(h1, 512, 0, 0, wcde2, P[35], vf, B, 8, 256, 2);
                    cde_stage_k<<<ewgrid(n), 256, 0, stream>>>(
                        vf, lat, Mb, times, zcur, kb[st],
                        (st < 3) ? ztmp : (float*)0,
                        n, itv, sub, fracs[st], coefs[st], F, CHW);
                }
                cde_rk4_k<<<ewgrid(n), 256, 0, stream>>>(
                    zcur, kb1, kb2, kb3, kb4, times, zall,
                    n, itv, (sub == 3) ? (itv + 1) : 0, F, CHW);
            }
        }
    }

    // ======================= up path ========================================
    {
        long n = (long)NF * 256 * 16 * 16;
        upsample2_k<<<ewgrid(n), 256, 0, stream>>>(zall, p2, n, 8, 8);
    }
    conv(p2, 256, res4, 256, wu1a, P[37], t4, NF, 16, 256, 1);
    conv(t4, 256, 0, 0, wu1b, P[39], res4, NF, 16, 256, 1);
    {
        long n = (long)NF * 128 * 32 * 32;
        convT2_k<<<ewgrid(n), 256, 0, stream>>>(res4, P[52], P[53], p1, n, 256, 128, 16, 16);
    }
    conv(p1, 128, res3, 128, wu2a, P[41], t3, NF, 32, 128, 1);
    conv(t3, 128, 0, 0, wu2b, P[43], res3, NF, 32, 128, 1);
    {
        long n = (long)NF * 64 * 64 * 64;
        convT2_k<<<ewgrid(n), 256, 0, stream>>>(res3, P[54], P[55], t2, n, 128, 64, 32, 32);
    }
    conv(t2, 64, res2, 64, wu3a, P[45], t1, NF, 64, 64, 1);
    conv(t1, 64, 0, 0, wu3b, P[47], res2, NF, 64, 64, 1);
    {
        long n = (long)NF * 32 * 128 * 128;
        convT2_k<<<ewgrid(n), 256, 0, stream>>>(res2, P[56], P[57], t1, n, 64, 32, 64, 64);
    }
    conv(t1, 32, res1, 32, wu4a, P[49], big, NF, 128, 32, 1);
    conv(big, 32, 0, 0, wu4b, P[51], res1, NF, 128, 32, 1);
    {
        long n = (long)NF * 128 * 128;
        conv1x1_sig_k<<<ewgrid(n), 256, 0, stream>>>(res1, P[58], P[59],
                                                     (float*)d_out, n, 32, 128 * 128);
    }
}
